// ReferenceLossAttnProcessor_90623809946240
// MI455X (gfx1250) — matrix operand order repaired and checked by reading, not compiled, not run on hardware
//
#include <hip/hip_runtime.h>
#include <hip/hip_bf16.h>
#include <math.h>

// ---------------------------------------------------------------------------
// MI455X / gfx1250 attention processor, bf16 WMMA pipeline, fp32 accumulation.
// Async global->LDS staging (ASYNCcnt) for contiguous tiles.
// ---------------------------------------------------------------------------

typedef __attribute__((ext_vector_type(16))) __bf16 v16bf;
typedef __attribute__((ext_vector_type(4)))  __bf16 bf16x4;
typedef __attribute__((ext_vector_type(8)))  float  v8f;

static constexpr int Bn = 2, Sn = 2048, Cn = 1280, Hn = 20, Dn = 64;
static constexpr int Mrows = Bn * Sn;              // 4096
static constexpr size_t N_HS = (size_t)Bn * Sn * Cn;  // 5,242,880
static constexpr size_t N_W  = (size_t)Cn * Cn;       // 1,638,400

// workspace byte offsets (all 256-aligned)
static constexpr size_t OFF_HS   = 0;
static constexpr size_t SZ_HS    = N_HS * 2;                 // 10,485,760
static constexpr size_t OFF_WQ   = OFF_HS + SZ_HS;
static constexpr size_t SZ_W     = N_W * 2;                  // 3,276,800
static constexpr size_t OFF_WK   = OFF_WQ + SZ_W;
static constexpr size_t OFF_WV   = OFF_WK + SZ_W;
static constexpr size_t OFF_WO   = OFF_WV + SZ_W;
static constexpr size_t OFF_Q    = OFF_WO + SZ_W;
static constexpr size_t OFF_K    = OFF_Q + SZ_HS;
static constexpr size_t OFF_V    = OFF_K + SZ_HS;
static constexpr size_t OFF_ATTN = OFF_V + SZ_HS;
static constexpr size_t OFF_PART = OFF_ATTN + SZ_HS;
static constexpr int NPART = 40 * 16 * 8;                    // 5120 partials

__device__ __forceinline__ v8f wmma_bf16(v16bf a, v16bf b, v8f c) {
  // D(16x16 f32) = A(16x32 bf16) x B(32x16 bf16) + C
  return __builtin_amdgcn_wmma_f32_16x16x32_bf16(
      /*neg_a=*/false, a, /*neg_b=*/false, b,
      /*c_mod=*/(short)0, c, /*reuse_a=*/false, /*reuse_b=*/false);
}

// CDNA5 async global->LDS copy, 16 bytes per lane, tracked by ASYNCcnt.
// LDS offset = low 32 bits of the generic shared-pointer (allocation-relative).
__device__ __forceinline__ void async_load_b128(unsigned lds_off, const void* gaddr) {
  asm volatile("global_load_async_to_lds_b128 %0, %1, off"
               :: "v"(lds_off), "v"(gaddr)
               : "memory");
}
__device__ __forceinline__ void wait_asynccnt0() {
  asm volatile("s_wait_asynccnt 0x0" ::: "memory");
}
__device__ __forceinline__ unsigned lds_off(const void* p) {
  return (unsigned)(uintptr_t)p;
}

// A fragment: 16x32 bf16 tile, row-major with given element stride.
// Lane (0..31): row = lane&15, lane-half h selects K range 16h..16h+15 (contiguous)
__device__ __forceinline__ v16bf load_frag_a32(const __bf16* tile, int stride, int lane) {
  const int r  = lane & 15;
  const int lh = (lane >> 4) & 1;
  const __bf16* row = tile + (size_t)r * stride;
  bf16x4 c0 = *(const bf16x4*)(row + 16 * lh);
  bf16x4 c1 = *(const bf16x4*)(row + 16 * lh + 4);
  bf16x4 c2 = *(const bf16x4*)(row + 16 * lh + 8);
  bf16x4 c3 = *(const bf16x4*)(row + 16 * lh + 12);
  v16bf f;
  f[0]=c0[0];  f[1]=c0[1];  f[2]=c0[2];  f[3]=c0[3];
  f[4]=c1[0];  f[5]=c1[1];  f[6]=c1[2];  f[7]=c1[3];
  f[8]=c2[0];  f[9]=c2[1];  f[10]=c2[2]; f[11]=c2[3];
  f[12]=c3[0]; f[13]=c3[1]; f[14]=c3[2]; f[15]=c3[3];
  return f;
}

// B fragment from a TRANSPOSED tile Bt[N=16][K=32] (row-major, stride elems).
// Lane: n = lane&15, half h selects K range 16h..16h+15 (contiguous).
__device__ __forceinline__ v16bf load_frag_b32(const __bf16* tileT, int stride, int lane) {
  const int n  = lane & 15;
  const int lh = (lane >> 4) & 1;
  const __bf16* row = tileT + (size_t)n * stride + 16 * lh;
  bf16x4 c0 = *(const bf16x4*)(row + 0);
  bf16x4 c1 = *(const bf16x4*)(row + 4);
  bf16x4 c2 = *(const bf16x4*)(row + 8);
  bf16x4 c3 = *(const bf16x4*)(row + 12);
  v16bf f;
  f[0]=c0[0];  f[1]=c0[1];  f[2]=c0[2];  f[3]=c0[3];
  f[4]=c1[0];  f[5]=c1[1];  f[6]=c1[2];  f[7]=c1[3];
  f[8]=c2[0];  f[9]=c2[1];  f[10]=c2[2]; f[11]=c2[3];
  f[12]=c3[0]; f[13]=c3[1]; f[14]=c3[2]; f[15]=c3[3];
  return f;
}

// ---------------------------------------------------------------------------
// fp32 -> bf16 conversion
// ---------------------------------------------------------------------------
__global__ __launch_bounds__(256) void f32_to_bf16_kernel(const float* __restrict__ src,
                                                          __bf16* __restrict__ dst, int n) {
  int i = (blockIdx.x * 256 + threadIdx.x) * 4;
  if (i + 3 < n) {
    float4 v = *(const float4*)(src + i);
    bf16x4 o;
    o[0] = (__bf16)v.x; o[1] = (__bf16)v.y; o[2] = (__bf16)v.z; o[3] = (__bf16)v.w;
    *(bf16x4*)(dst + i) = o;
  }
}

// ---------------------------------------------------------------------------
// Fused QKV GEMM: [4096,1280] x [1280,1280] (z selects Wq/Wk/Wv),
// output written bf16 in head-split [B,H,S,D] layout.
// Block: 256 thr, tile 128(M) x 128(N), 8 waves as 2x4, wave tile 64x32.
// A tile staged via async global->LDS; B tile transposed via b64 loads.
// ---------------------------------------------------------------------------
__global__ __launch_bounds__(256) void gemm_qkv_kernel(
    const __bf16* __restrict__ X, const __bf16* __restrict__ Wqb,
    const __bf16* __restrict__ Wkb, const __bf16* __restrict__ Wvb,
    __bf16* __restrict__ Qo, __bf16* __restrict__ Ko, __bf16* __restrict__ Vo) {
  __shared__ __bf16 Al[128 * 40];   // A tile 128x32, stride 40
  __shared__ __bf16 Btl[128 * 40];  // B tile transposed: [n=128][k=32], stride 40

  const int z = blockIdx.z;
  const __bf16* W = (z == 0) ? Wqb : (z == 1) ? Wkb : Wvb;
  __bf16* O = (z == 0) ? Qo : (z == 1) ? Ko : Vo;

  const int n0 = blockIdx.x * 128;
  const int m0 = blockIdx.y * 128;
  const int tid = threadIdx.x;
  const int lane = tid & 31;
  const int wid = tid >> 5;        // 0..7
  const int wr = wid >> 2;         // 0..1 -> rows 64*wr
  const int wc = wid & 3;          // 0..3 -> cols 32*wc

  v8f acc[4][2] = {};

  const int arow = tid >> 1, aseg = tid & 1;         // A staging: 16 bf16/thread
  const int bn4 = (tid & 31) * 4, bkk = (tid >> 5) * 4;  // B staging: 4x b64/thread
  const unsigned a_lds = lds_off(Al + arow * 40 + 16 * aseg);

  for (int k0 = 0; k0 < Cn; k0 += 32) {
    // stage A: rows m0..m0+127, cols k0..k0+31 (two async b128 per thread)
    {
      const __bf16* a_g = X + (size_t)(m0 + arow) * Cn + k0 + 16 * aseg;
      async_load_b128(a_lds, a_g);
      async_load_b128(a_lds + 16, a_g + 8);
    }
    // stage B transposed: Btl[n][kk] = W[k0+kk][n0+n]  (coalesced b64 reads)
#pragma unroll
    for (int i = 0; i < 4; ++i) {
      bf16x4 w = *(const bf16x4*)(W + (size_t)(k0 + bkk + i) * Cn + n0 + bn4);
#pragma unroll
      for (int j = 0; j < 4; ++j)
        Btl[(bn4 + j) * 40 + bkk + i] = w[j];
    }
    wait_asynccnt0();
    __syncthreads();

    v16bf bfr[2];
#pragma unroll
    for (int j = 0; j < 2; ++j)
      bfr[j] = load_frag_b32(Btl + (32 * wc + 16 * j) * 40, 40, lane);
#pragma unroll
    for (int i = 0; i < 4; ++i) {
      v16bf afr = load_frag_a32(Al + (64 * wr + 16 * i) * 40, 40, lane);
#pragma unroll
      for (int j = 0; j < 2; ++j)
        acc[i][j] = wmma_bf16(afr, bfr[j], acc[i][j]);
    }
    __syncthreads();
  }

  // epilogue: write bf16 into [B,H,S,D]
  const int lh = lane >> 4, nl = lane & 15;
#pragma unroll
  for (int i = 0; i < 4; ++i) {
#pragma unroll
    for (int j = 0; j < 2; ++j) {
#pragma unroll
      for (int r = 0; r < 8; ++r) {
        int m = m0 + 64 * wr + 16 * i + r + 8 * lh;
        int n = n0 + 32 * wc + 16 * j + nl;
        int bb = m >> 11, s = m & (Sn - 1);
        int hh = n >> 6, d = n & (Dn - 1);
        O[((size_t)(bb * Hn + hh) * Sn + s) * Dn + d] = (__bf16)acc[i][j][r];
      }
    }
  }
}

// ---------------------------------------------------------------------------
// Output projection: attn[4096,1280] x Wo + bo + residual -> fp32 d_out
// ---------------------------------------------------------------------------
__global__ __launch_bounds__(256) void gemm_out_kernel(
    const __bf16* __restrict__ A, const __bf16* __restrict__ Wob,
    const float* __restrict__ bo, const float* __restrict__ resid,
    float* __restrict__ out) {
  __shared__ __bf16 Al[128 * 40];
  __shared__ __bf16 Btl[128 * 40];

  const int n0 = blockIdx.x * 128;
  const int m0 = blockIdx.y * 128;
  const int tid = threadIdx.x;
  const int lane = tid & 31;
  const int wid = tid >> 5;
  const int wr = wid >> 2, wc = wid & 3;

  v8f acc[4][2] = {};
  const int arow = tid >> 1, aseg = tid & 1;
  const int bn4 = (tid & 31) * 4, bkk = (tid >> 5) * 4;
  const unsigned a_lds = lds_off(Al + arow * 40 + 16 * aseg);

  for (int k0 = 0; k0 < Cn; k0 += 32) {
    {
      const __bf16* a_g = A + (size_t)(m0 + arow) * Cn + k0 + 16 * aseg;
      async_load_b128(a_lds, a_g);
      async_load_b128(a_lds + 16, a_g + 8);
    }
#pragma unroll
    for (int i = 0; i < 4; ++i) {
      bf16x4 w = *(const bf16x4*)(Wob + (size_t)(k0 + bkk + i) * Cn + n0 + bn4);
#pragma unroll
      for (int j = 0; j < 4; ++j)
        Btl[(bn4 + j) * 40 + bkk + i] = w[j];
    }
    wait_asynccnt0();
    __syncthreads();

    v16bf bfr[2];
#pragma unroll
    for (int j = 0; j < 2; ++j)
      bfr[j] = load_frag_b32(Btl + (32 * wc + 16 * j) * 40, 40, lane);
#pragma unroll
    for (int i = 0; i < 4; ++i) {
      v16bf afr = load_frag_a32(Al + (64 * wr + 16 * i) * 40, 40, lane);
#pragma unroll
      for (int j = 0; j < 2; ++j)
        acc[i][j] = wmma_bf16(afr, bfr[j], acc[i][j]);
    }
    __syncthreads();
  }

  const int lh = lane >> 4, nl = lane & 15;
#pragma unroll
  for (int i = 0; i < 4; ++i) {
#pragma unroll
    for (int j = 0; j < 2; ++j) {
#pragma unroll
      for (int r = 0; r < 8; ++r) {
        int m = m0 + 64 * wr + 16 * i + r + 8 * lh;
        int n = n0 + 32 * wc + 16 * j + nl;
        size_t idx = (size_t)m * Cn + n;
        out[idx] = acc[i][j][r] + bo[n] + resid[idx];
      }
    }
  }
}

// ---------------------------------------------------------------------------
// Flash attention per (b,h): online softmax + Frobenius-sum of probs.
// Block = 256 thr = 8 waves; block covers 128 q rows; wave owns 16 q rows.
// KV streamed in steps of 32 keys through LDS (K async as-is, V transposed).
// ---------------------------------------------------------------------------
__global__ __launch_bounds__(256) void attn_kernel(
    const __bf16* __restrict__ Qb, const __bf16* __restrict__ Kb,
    const __bf16* __restrict__ Vb, __bf16* __restrict__ attnout,
    float* __restrict__ partials) {
  __shared__ __bf16 Kl[32 * 72];        // [key][d], stride 72
  __shared__ __bf16 Vt[64 * 40];        // [d][key], stride 40
  __shared__ __bf16 Pl[8 * 16 * 40];    // per-wave P tile [16][32], stride 40

  const int bh = blockIdx.x;            // 0..39  (b*H + h)
  const int q0 = blockIdx.y * 128;
  const int bb = bh / Hn, hh = bh % Hn;
  const int tid = threadIdx.x, lane = tid & 31, wid = tid >> 5;
  const int lh = lane >> 4, nl = lane & 15;

  // resident Q fragments (16 rows x 64 d -> two K=32 chunks)
  const __bf16* qbase = Qb + ((size_t)bh * Sn + q0 + wid * 16) * Dn;
  v16bf qf0 = load_frag_a32(qbase, Dn, lane);
  v16bf qf1 = load_frag_a32(qbase + 32, Dn, lane);

  v8f oacc[4] = {};
  float rm[8], rl[8], rsq[8];
#pragma unroll
  for (int r = 0; r < 8; ++r) { rm[r] = -1e30f; rl[r] = 0.f; rsq[r] = 0.f; }

  const int skey = tid >> 3;            // 0..31
  const int sd0 = (tid & 7) * 8;        // 0..56
  const __bf16* ksrc = Kb + (size_t)bh * Sn * Dn;
  const __bf16* vsrc = Vb + (size_t)bh * Sn * Dn;
  const unsigned k_lds = lds_off(Kl + skey * 72 + sd0);
  const float scale = 0.125f;           // 1/sqrt(64)

  for (int kv0 = 0; kv0 < Sn; kv0 += 32) {
    // stage K tile [32 keys][64 d] via async copy (rows of k == Bt layout)
    async_load_b128(k_lds, ksrc + (size_t)(kv0 + skey) * Dn + sd0);
    // stage V transposed: Vt[d][key]
    {
      bf16x4 v0 = *(const bf16x4*)(vsrc + (size_t)(kv0 + skey) * Dn + sd0);
      bf16x4 v1 = *(const bf16x4*)(vsrc + (size_t)(kv0 + skey) * Dn + sd0 + 4);
#pragma unroll
      for (int i = 0; i < 4; ++i) Vt[(sd0 + i) * 40 + skey] = v0[i];
#pragma unroll
      for (int i = 0; i < 4; ++i) Vt[(sd0 + 4 + i) * 40 + skey] = v1[i];
    }
    wait_asynccnt0();
    __syncthreads();

    // scores: two 16x16 tiles (keys kv0..+15, kv0+16..+31), K-dim = D = 2x32
    v8f s0 = {}, s1 = {};
    {
      v16bf b00 = load_frag_b32(Kl + 0 * 72, 72, lane);
      v16bf b01 = load_frag_b32(Kl + 0 * 72 + 32, 72, lane);
      s0 = wmma_bf16(qf0, b00, s0);
      s0 = wmma_bf16(qf1, b01, s0);
      v16bf b10 = load_frag_b32(Kl + 16 * 72, 72, lane);
      v16bf b11 = load_frag_b32(Kl + 16 * 72 + 32, 72, lane);
      s1 = wmma_bf16(qf0, b10, s1);
      s1 = wmma_bf16(qf1, b11, s1);
    }

    // online softmax update (row = r + 8*lh, 16 lanes per row; xor masks<16
    // keep the butterfly within each half-wave = one score row)
    float alpha[8];
#pragma unroll
    for (int r = 0; r < 8; ++r) {
      float x0 = s0[r] * scale, x1 = s1[r] * scale;
      float mx = fmaxf(x0, x1);
      mx = fmaxf(mx, __shfl_xor(mx, 1, 32));
      mx = fmaxf(mx, __shfl_xor(mx, 2, 32));
      mx = fmaxf(mx, __shfl_xor(mx, 4, 32));
      mx = fmaxf(mx, __shfl_xor(mx, 8, 32));
      float mnew = fmaxf(rm[r], mx);
      float p0 = __expf(x0 - mnew), p1 = __expf(x1 - mnew);
      float ps = p0 + p1;
      float pq = p0 * p0 + p1 * p1;
      ps += __shfl_xor(ps, 1, 32); pq += __shfl_xor(pq, 1, 32);
      ps += __shfl_xor(ps, 2, 32); pq += __shfl_xor(pq, 2, 32);
      ps += __shfl_xor(ps, 4, 32); pq += __shfl_xor(pq, 4, 32);
      ps += __shfl_xor(ps, 8, 32); pq += __shfl_xor(pq, 8, 32);
      float a = __expf(rm[r] - mnew);
      rl[r] = rl[r] * a + ps;
      rsq[r] = rsq[r] * a * a + pq;
      rm[r] = mnew;
      alpha[r] = a;
      s0[r] = p0;
      s1[r] = p1;
    }

    // rescale O accumulators
#pragma unroll
    for (int t = 0; t < 4; ++t)
#pragma unroll
      for (int r = 0; r < 8; ++r) oacc[t][r] *= alpha[r];

    // re-stripe P (C-layout -> A-fragment) through this wave's LDS region
    __bf16* Pw = Pl + wid * 16 * 40;
#pragma unroll
    for (int r = 0; r < 8; ++r) {
      int row = r + 8 * lh;
      Pw[row * 40 + nl] = (__bf16)s0[r];
      Pw[row * 40 + 16 + nl] = (__bf16)s1[r];
    }
    v16bf pf = load_frag_a32(Pw, 40, lane);   // P 16x32 (K = 32 keys)

    // O += P @ V  (four 16-wide d tiles)
#pragma unroll
    for (int t = 0; t < 4; ++t) {
      v16bf vf = load_frag_b32(Vt + 16 * t * 40, 40, lane);
      oacc[t] = wmma_bf16(pf, vf, oacc[t]);
    }
    __syncthreads();
  }

  // write O/l as bf16 into [B,S,C] for the output projection
#pragma unroll
  for (int r = 0; r < 8; ++r) {
    int s = q0 + wid * 16 + r + 8 * lh;
    size_t base = ((size_t)bb * Sn + s) * Cn + hh * Dn;
    float invl = 1.0f / rl[r];
#pragma unroll
    for (int t = 0; t < 4; ++t)
      attnout[base + 16 * t + nl] = (__bf16)(oacc[t][r] * invl);
  }

  // deterministic loss partial: one value per (block, wave)
  float part = 0.f;
  if (nl == 0) {
#pragma unroll
    for (int r = 0; r < 8; ++r) part += rsq[r] / (rl[r] * rl[r]);
  }
  part += __shfl_xor(part, 16, 32);   // combine rows 0-7 with rows 8-15
  if (lane == 0)
    partials[(blockIdx.y * 40 + blockIdx.x) * 8 + wid] = part;
}

// ---------------------------------------------------------------------------
// Fixed-order reduction of loss partials -> sqrt -> d_out[B*S*C]
// ---------------------------------------------------------------------------
__global__ __launch_bounds__(256) void loss_reduce_kernel(const float* __restrict__ partials,
                                                          float* __restrict__ loss_out) {
  __shared__ float red[256];
  float v = 0.f;
  for (int i = threadIdx.x; i < NPART; i += 256) v += partials[i];
  red[threadIdx.x] = v;
  __syncthreads();
  for (int s = 128; s > 0; s >>= 1) {
    if (threadIdx.x < s) red[threadIdx.x] += red[threadIdx.x + s];
    __syncthreads();
  }
  if (threadIdx.x == 0) *loss_out = sqrtf(red[0]);
}

// ---------------------------------------------------------------------------
extern "C" void kernel_launch(void* const* d_in, const int* in_sizes, int n_in,
                              void* d_out, int out_size, void* d_ws, size_t ws_size,
                              hipStream_t stream) {
  const float* hs = (const float*)d_in[0];
  const float* Wq = (const float*)d_in[1];
  const float* Wk = (const float*)d_in[2];
  const float* Wv = (const float*)d_in[3];
  const float* Wo = (const float*)d_in[4];
  const float* bo = (const float*)d_in[5];
  float* out = (float*)d_out;

  char* ws = (char*)d_ws;
  __bf16* hsb = (__bf16*)(ws + OFF_HS);
  __bf16* wqb = (__bf16*)(ws + OFF_WQ);
  __bf16* wkb = (__bf16*)(ws + OFF_WK);
  __bf16* wvb = (__bf16*)(ws + OFF_WV);
  __bf16* wob = (__bf16*)(ws + OFF_WO);
  __bf16* qb  = (__bf16*)(ws + OFF_Q);
  __bf16* kb  = (__bf16*)(ws + OFF_K);
  __bf16* vb  = (__bf16*)(ws + OFF_V);
  __bf16* ab  = (__bf16*)(ws + OFF_ATTN);
  float* partials = (float*)(ws + OFF_PART);

  // fp32 -> bf16 conversions
  f32_to_bf16_kernel<<<(int)(N_HS / 1024), 256, 0, stream>>>(hs, hsb, (int)N_HS);
  f32_to_bf16_kernel<<<(int)(N_W / 1024), 256, 0, stream>>>(Wq, wqb, (int)N_W);
  f32_to_bf16_kernel<<<(int)(N_W / 1024), 256, 0, stream>>>(Wk, wkb, (int)N_W);
  f32_to_bf16_kernel<<<(int)(N_W / 1024), 256, 0, stream>>>(Wv, wvb, (int)N_W);
  f32_to_bf16_kernel<<<(int)(N_W / 1024), 256, 0, stream>>>(Wo, wob, (int)N_W);

  // fused QKV projections (z selects matrix), head-split bf16 outputs
  gemm_qkv_kernel<<<dim3(Cn / 128, Mrows / 128, 3), 256, 0, stream>>>(
      hsb, wqb, wkb, wvb, qb, kb, vb);

  // flash attention + Frobenius loss partials
  attn_kernel<<<dim3(Bn * Hn, Sn / 128), 256, 0, stream>>>(qb, kb, vb, ab, partials);

  // output projection + bias + residual (fp32 out)
  gemm_out_kernel<<<dim3(Cn / 128, Mrows / 128), 256, 0, stream>>>(ab, wob, bo, hs, out);

  // loss = sqrt(sum of partials), appended after the [B,S,C] output
  loss_reduce_kernel<<<1, 256, 0, stream>>>(partials, out + N_HS);
}
